// SpatialNCA_29566554865725
// MI455X (gfx1250) — matrix operand order repaired and checked by reading, not compiled, not run on hardware
//
#include <hip/hip_runtime.h>
#include <math.h>

// ---------------- types ----------------
typedef __bf16 bf16_t;
typedef bf16_t v16bf __attribute__((ext_vector_type(16)));
typedef bf16_t v8bf  __attribute__((ext_vector_type(8)));
typedef float  v8f   __attribute__((ext_vector_type(8)));

union BF16x16 { v16bf v; v8bf h[2]; };

#define NNODES 50000
#define NEDGES 500000
#define ETILES (NEDGES / 16)      // 31250 wave tiles of 16 edges
#define NODETILES (NNODES / 16)   // 3125 wave tiles of 16 nodes
#define WPB 8                     // waves per 256-thread block (wave32)

__device__ inline void atomAddG(float* p, float v) {
  __hip_atomic_fetch_add(p, v, __ATOMIC_RELAXED, __HIP_MEMORY_SCOPE_AGENT);
}
__device__ inline void atomAddL(float* p, float v) {
  __hip_atomic_fetch_add(p, v, __ATOMIC_RELAXED, __HIP_MEMORY_SCOPE_WORKGROUP);
}

// BN(alpha,beta)+ReLU applied elementwise to an 8-wide bf16 chunk at column kc.
__device__ inline v8bf bn_relu8(v8bf x, const float* __restrict__ al,
                                const float* __restrict__ be, int kc) {
  v8bf r;
#pragma unroll
  for (int i = 0; i < 8; ++i) {
    float f = (float)x[i] * al[kc + i] + be[kc + i];
    f = f > 0.f ? f : 0.f;
    r[i] = (bf16_t)f;
  }
  return r;
}

// ---------------- small utility kernels ----------------
__global__ void zero_kernel(float* __restrict__ p, long n) {
  long i = (long)blockIdx.x * blockDim.x + threadIdx.x;
  long stride = (long)gridDim.x * blockDim.x;
  for (; i < n; i += stride) p[i] = 0.f;
}

__global__ void cvt_bf16_kernel(const float* __restrict__ in, bf16_t* __restrict__ out, long n) {
  long i = (long)blockIdx.x * blockDim.x + threadIdx.x;
  if (i < n) out[i] = (bf16_t)in[i];
}

// W row-major [K][Ncols] fp32 -> WT [Ncols][K] bf16 (so B-frag loads are contiguous per lane)
__global__ void transpose_cvt_kernel(const float* __restrict__ W, bf16_t* __restrict__ WT,
                                     int K, int Ncols) {
  int i = blockIdx.x * blockDim.x + threadIdx.x;
  int total = K * Ncols;
  if (i >= total) return;
  int k = i / Ncols, n = i % Ncols;
  WT[(size_t)n * K + k] = (bf16_t)W[i];
}

// alpha = g*rsqrt(var+eps); beta = b - mean*alpha  (biased variance, training-mode BN)
__global__ void bn_finalize_kernel(const float* __restrict__ stats, const float* __restrict__ g,
                                   const float* __restrict__ b, float cntN, int C,
                                   float* __restrict__ alpha, float* __restrict__ beta) {
  int i = blockIdx.x * blockDim.x + threadIdx.x;
  if (i < C) {
    float m = stats[i] / cntN;
    float v = stats[C + i] / cntN - m * m;
    float a = g[i] * rsqrtf(v + 1e-5f);
    alpha[i] = a;
    beta[i]  = b[i] - m * a;
  }
}

// ---------------- GEMM 1: [h_i | h_j | dist] @ Wm1 + bm1 ----------------
__global__ __launch_bounds__(256) void edge_gemm1_kernel(
    const bf16_t* __restrict__ hbf, const int* __restrict__ src, const int* __restrict__ dst,
    const float* __restrict__ pos, const bf16_t* __restrict__ WT1 /*[128][256]*/,
    const float* __restrict__ w256row /*Wm1 row 256*/, const float* __restrict__ bias,
    bf16_t* __restrict__ Aout, float* __restrict__ stats, int tiles) {
  __shared__ float sred[256];
  int tid = threadIdx.x, lane = tid & 31, wslot = tid >> 5;
  int tile = blockIdx.x * WPB + wslot;
  sred[tid] = 0.f;
  __syncthreads();

  if (tile < tiles) {
    int r = lane & 15, kh = lane >> 4;
    size_t e = (size_t)tile * 16 + r;
    int si = src[e], di = dst[e];

    float dx = pos[2 * si] - pos[2 * di];
    float dy = pos[2 * si + 1] - pos[2 * di + 1];
    float dist = sqrtf(dx * dx + dy * dy);

    v8f acc[8];
#pragma unroll
    for (int t = 0; t < 8; ++t) {
      float bv = bias[16 * t + r];
#pragma unroll
      for (int v = 0; v < 8; ++v) acc[t][v] = bv;
    }

    // rank-1 update for the dist column (K=256): co-executes with WMMA
    float w256[8];
#pragma unroll
    for (int t = 0; t < 8; ++t) w256[t] = w256row[16 * t + r];
#pragma unroll
    for (int v = 0; v < 8; ++v) {
      float dv = __shfl(dist, v + 8 * kh, 32);
#pragma unroll
      for (int t = 0; t < 8; ++t) acc[t][v] += dv * w256[t];
    }

#pragma unroll
    for (int kt = 0; kt < 8; ++kt) {
      // K 0..127 = h[dst] (h_i), K 128..255 = h[src] (h_j)
      const bf16_t* ab = (kt < 4) ? (hbf + (size_t)di * 128 + kt * 32)
                                  : (hbf + (size_t)si * 128 + (kt - 4) * 32);
      BF16x16 a;
      a.h[0] = *(const v8bf*)(ab + 8 * kh);
      a.h[1] = *(const v8bf*)(ab + 16 + 8 * kh);
#pragma unroll
      for (int t = 0; t < 8; ++t) {
        const bf16_t* bb = WT1 + (size_t)(16 * t + r) * 256 + kt * 32 + 8 * kh;
        BF16x16 b;
        b.h[0] = *(const v8bf*)bb;
        b.h[1] = *(const v8bf*)(bb + 16);
        acc[t] = __builtin_amdgcn_wmma_f32_16x16x32_bf16(false, a.v, false, b.v,
                                                         (short)0, acc[t], false, false);
      }
    }

#pragma unroll
    for (int t = 0; t < 8; ++t) {
      float s = 0.f, ss = 0.f;
#pragma unroll
      for (int v = 0; v < 8; ++v) {
        float z = acc[t][v];
        s += z; ss += z * z;
        size_t row = (size_t)tile * 16 + v + 8 * kh;
        Aout[row * 128 + 16 * t + r] = (bf16_t)z;
      }
      atomAddL(&sred[16 * t + r], s);
      atomAddL(&sred[128 + 16 * t + r], ss);
    }
  }
  __syncthreads();
  atomAddG(&stats[tid], sred[tid]);
}

// ---------------- GEMM 2: bnrelu(A1) @ Wm2 + bm2 ----------------
__global__ __launch_bounds__(256) void edge_gemm2_kernel(
    const bf16_t* __restrict__ Ain, const float* __restrict__ al, const float* __restrict__ be,
    const bf16_t* __restrict__ WT /*[128][128]*/, const float* __restrict__ bias,
    bf16_t* __restrict__ Aout, float* __restrict__ stats, int tiles) {
  __shared__ float sred[256];
  int tid = threadIdx.x, lane = tid & 31, wslot = tid >> 5;
  int tile = blockIdx.x * WPB + wslot;
  sred[tid] = 0.f;
  __syncthreads();

  if (tile < tiles) {
    int r = lane & 15, kh = lane >> 4;
    size_t e = (size_t)tile * 16 + r;
    v8f acc[8];
#pragma unroll
    for (int t = 0; t < 8; ++t) {
      float bv = bias[16 * t + r];
#pragma unroll
      for (int v = 0; v < 8; ++v) acc[t][v] = bv;
    }

#pragma unroll
    for (int kt = 0; kt < 4; ++kt) {
      const bf16_t* ab = Ain + e * 128 + kt * 32;
      BF16x16 a;
      a.h[0] = bn_relu8(*(const v8bf*)(ab + 8 * kh), al, be, kt * 32 + 8 * kh);
      a.h[1] = bn_relu8(*(const v8bf*)(ab + 16 + 8 * kh), al, be, kt * 32 + 16 + 8 * kh);
#pragma unroll
      for (int t = 0; t < 8; ++t) {
        const bf16_t* bb = WT + (size_t)(16 * t + r) * 128 + kt * 32 + 8 * kh;
        BF16x16 b;
        b.h[0] = *(const v8bf*)bb;
        b.h[1] = *(const v8bf*)(bb + 16);
        acc[t] = __builtin_amdgcn_wmma_f32_16x16x32_bf16(false, a.v, false, b.v,
                                                         (short)0, acc[t], false, false);
      }
    }

#pragma unroll
    for (int t = 0; t < 8; ++t) {
      float s = 0.f, ss = 0.f;
#pragma unroll
      for (int v = 0; v < 8; ++v) {
        float z = acc[t][v];
        s += z; ss += z * z;
        size_t row = (size_t)tile * 16 + v + 8 * kh;
        Aout[row * 128 + 16 * t + r] = (bf16_t)z;
      }
      atomAddL(&sred[16 * t + r], s);
      atomAddL(&sred[128 + 16 * t + r], ss);
    }
  }
  __syncthreads();
  atomAddG(&stats[tid], sred[tid]);
}

// ---------------- GEMM 3: msg=bnrelu(pre); scatter msg->h_aggr; msg @ Wp1 + bp1 ----------------
__global__ __launch_bounds__(256) void edge_gemm3_kernel(
    const bf16_t* __restrict__ Ain, const float* __restrict__ al, const float* __restrict__ be,
    const bf16_t* __restrict__ WT /*[128][128]*/, const float* __restrict__ bias,
    const int* __restrict__ dst, float* __restrict__ h_aggr, float* __restrict__ cnt,
    bf16_t* __restrict__ Aout, float* __restrict__ stats, int tiles) {
  __shared__ float sred[256];
  int tid = threadIdx.x, lane = tid & 31, wslot = tid >> 5;
  int tile = blockIdx.x * WPB + wslot;
  sred[tid] = 0.f;
  __syncthreads();

  if (tile < tiles) {
    int r = lane & 15, kh = lane >> 4;
    size_t e = (size_t)tile * 16 + r;
    int di = dst[e];
    if (lane < 16) atomAddG(&cnt[di], 1.0f);  // one count per edge

    v8f acc[8];
#pragma unroll
    for (int t = 0; t < 8; ++t) {
      float bv = bias[16 * t + r];
#pragma unroll
      for (int v = 0; v < 8; ++v) acc[t][v] = bv;
    }

#pragma unroll
    for (int kt = 0; kt < 4; ++kt) {
      BF16x16 a;
#pragma unroll
      for (int c = 0; c < 2; ++c) {
        int kc = kt * 32 + 8 * kh + 16 * c;  // each (e,k) visited exactly once across the wave
        v8bf raw = *(const v8bf*)(Ain + e * 128 + kc);
        v8bf outh;
#pragma unroll
        for (int i = 0; i < 8; ++i) {
          float f = (float)raw[i] * al[kc + i] + be[kc + i];
          f = f > 0.f ? f : 0.f;
          atomAddG(&h_aggr[(size_t)di * 128 + kc + i], f);  // segment-sum of msg
          outh[i] = (bf16_t)f;
        }
        a.h[c] = outh;
      }
#pragma unroll
      for (int t = 0; t < 8; ++t) {
        const bf16_t* bb = WT + (size_t)(16 * t + r) * 128 + kt * 32 + 8 * kh;
        BF16x16 b;
        b.h[0] = *(const v8bf*)bb;
        b.h[1] = *(const v8bf*)(bb + 16);
        acc[t] = __builtin_amdgcn_wmma_f32_16x16x32_bf16(false, a.v, false, b.v,
                                                         (short)0, acc[t], false, false);
      }
    }

#pragma unroll
    for (int t = 0; t < 8; ++t) {
      float s = 0.f, ss = 0.f;
#pragma unroll
      for (int v = 0; v < 8; ++v) {
        float z = acc[t][v];
        s += z; ss += z * z;
        size_t row = (size_t)tile * 16 + v + 8 * kh;
        Aout[row * 128 + 16 * t + r] = (bf16_t)z;
      }
      atomAddL(&sred[16 * t + r], s);
      atomAddL(&sred[128 + 16 * t + r], ss);
    }
  }
  __syncthreads();
  atomAddG(&stats[tid], sred[tid]);
}

// ---------------- node GEMM: [h | h_aggr] @ Wu1 + bu1 ----------------
__global__ __launch_bounds__(256) void node_gemm_kernel(
    const bf16_t* __restrict__ hbf, const float* __restrict__ hag,
    const bf16_t* __restrict__ WT /*[128][256]*/, const float* __restrict__ bias,
    bf16_t* __restrict__ Uout, float* __restrict__ stats, int tiles) {
  __shared__ float sred[256];
  int tid = threadIdx.x, lane = tid & 31, wslot = tid >> 5;
  int tile = blockIdx.x * WPB + wslot;
  sred[tid] = 0.f;
  __syncthreads();

  if (tile < tiles) {
    int r = lane & 15, kh = lane >> 4;
    size_t n = (size_t)tile * 16 + r;
    v8f acc[8];
#pragma unroll
    for (int t = 0; t < 8; ++t) {
      float bv = bias[16 * t + r];
#pragma unroll
      for (int v = 0; v < 8; ++v) acc[t][v] = bv;
    }

#pragma unroll
    for (int kt = 0; kt < 8; ++kt) {
      BF16x16 a;
      if (kt < 4) {
        const bf16_t* ab = hbf + n * 128 + kt * 32;
        a.h[0] = *(const v8bf*)(ab + 8 * kh);
        a.h[1] = *(const v8bf*)(ab + 16 + 8 * kh);
      } else {
        const float* af = hag + n * 128 + (kt - 4) * 32;
#pragma unroll
        for (int c = 0; c < 2; ++c) {
          const float* p = af + 8 * kh + 16 * c;
          v8bf outh;
#pragma unroll
          for (int i = 0; i < 8; ++i) outh[i] = (bf16_t)p[i];
          a.h[c] = outh;
        }
      }
#pragma unroll
      for (int t = 0; t < 8; ++t) {
        const bf16_t* bb = WT + (size_t)(16 * t + r) * 256 + kt * 32 + 8 * kh;
        BF16x16 b;
        b.h[0] = *(const v8bf*)bb;
        b.h[1] = *(const v8bf*)(bb + 16);
        acc[t] = __builtin_amdgcn_wmma_f32_16x16x32_bf16(false, a.v, false, b.v,
                                                         (short)0, acc[t], false, false);
      }
    }

#pragma unroll
    for (int t = 0; t < 8; ++t) {
      float s = 0.f, ss = 0.f;
#pragma unroll
      for (int v = 0; v < 8; ++v) {
        float z = acc[t][v];
        s += z; ss += z * z;
        size_t row = (size_t)tile * 16 + v + 8 * kh;
        Uout[row * 128 + 16 * t + r] = (bf16_t)z;
      }
      atomAddL(&sred[16 * t + r], s);
      atomAddL(&sred[128 + 16 * t + r], ss);
    }
  }
  __syncthreads();
  atomAddG(&stats[tid], sred[tid]);
}

// ---------------- Wp2 matvec: t = bnrelu3(S1) . Wp2 + bp2, with scalar batch stats ----------------
__global__ __launch_bounds__(256) void matvec_p2_kernel(
    const bf16_t* __restrict__ S1, const float* __restrict__ al, const float* __restrict__ be,
    const float* __restrict__ Wp2, const float* __restrict__ bp2,
    float* __restrict__ tpre, float* __restrict__ stats2, int E) {
  __shared__ float sred[2];
  int tid = threadIdx.x;
  int e = blockIdx.x * 256 + tid;
  if (tid < 2) sred[tid] = 0.f;
  __syncthreads();

  float acc = 0.f;
  if (e < E) {
    acc = bp2[0];
    const bf16_t* row = S1 + (size_t)e * 128;
#pragma unroll
    for (int kk = 0; kk < 16; ++kk) {
      v8bf chunk = *(const v8bf*)(row + kk * 8);
#pragma unroll
      for (int i = 0; i < 8; ++i) {
        int k = kk * 8 + i;
        float f = (float)chunk[i] * al[k] + be[k];
        f = f > 0.f ? f : 0.f;
        acc += f * Wp2[k];
      }
    }
    tpre[e] = acc;
  }
  float s = acc, ss = acc * acc;  // zero for inactive threads
#pragma unroll
  for (int o = 16; o > 0; o >>= 1) {
    s += __shfl_down(s, o, 32);
    ss += __shfl_down(ss, o, 32);
  }
  if ((tid & 31) == 0) { atomAddL(&sred[0], s); atomAddL(&sred[1], ss); }
  __syncthreads();
  if (tid == 0) { atomAddG(&stats2[0], sred[0]); atomAddG(&stats2[1], sred[1]); }
}

// ---------------- pos scatter: x_sum[dst] += (pos[src]-pos[dst]) * scale ----------------
__global__ void pos_scatter_kernel(const float* __restrict__ tpre, const float* __restrict__ ab4,
                                   const int* __restrict__ src, const int* __restrict__ dst,
                                   const float* __restrict__ pos, float* __restrict__ xsum, int E) {
  int e = blockIdx.x * blockDim.x + threadIdx.x;
  if (e >= E) return;
  float sc = ab4[0] * tpre[e] + ab4[1];
  sc = sc > 0.f ? sc : 0.f;
  int s = src[e], d = dst[e];
  atomAddG(&xsum[2 * d],     (pos[2 * s]     - pos[2 * d])     * sc);
  atomAddG(&xsum[2 * d + 1], (pos[2 * s + 1] - pos[2 * d + 1]) * sc);
}

// ---------------- residual outputs ----------------
__global__ void final_h_kernel(const float* __restrict__ h, const bf16_t* __restrict__ U,
                               const float* __restrict__ al, const float* __restrict__ be,
                               float* __restrict__ out, long n) {
  long i = (long)blockIdx.x * blockDim.x + threadIdx.x;
  if (i >= n) return;
  int k = (int)(i & 127);
  float u = (float)U[i] * al[k] + be[k];
  u = u > 0.f ? u : 0.f;
  out[i] = h[i] + u;
}

__global__ void final_pos_kernel(const float* __restrict__ pos, const float* __restrict__ xsum,
                                 const float* __restrict__ cnt, float* __restrict__ out, int n2) {
  int i = blockIdx.x * blockDim.x + threadIdx.x;
  if (i >= n2) return;
  float c = cnt[i >> 1];
  c = c > 1.f ? c : 1.f;
  out[i] = pos[i] + xsum[i] / c;
}

// ---------------- host side ----------------
extern "C" void kernel_launch(void* const* d_in, const int* in_sizes, int n_in,
                              void* d_out, int out_size, void* d_ws, size_t ws_size,
                              hipStream_t stream) {
  (void)in_sizes; (void)n_in; (void)out_size; (void)ws_size;
  const float* h    = (const float*)d_in[0];
  const float* pos  = (const float*)d_in[1];
  const int*   ei   = (const int*)d_in[3];     // h_init (d_in[2]) unused: reinit=False
  const float* Wm1  = (const float*)d_in[4];
  const float* bm1  = (const float*)d_in[5];
  const float* gm1  = (const float*)d_in[6];
  const float* bbm1 = (const float*)d_in[7];
  const float* Wm2  = (const float*)d_in[8];
  const float* bm2  = (const float*)d_in[9];
  const float* gm2  = (const float*)d_in[10];
  const float* bbm2 = (const float*)d_in[11];
  const float* Wp1  = (const float*)d_in[12];
  const float* bp1  = (const float*)d_in[13];
  const float* gp1  = (const float*)d_in[14];
  const float* bbp1 = (const float*)d_in[15];
  const float* Wp2  = (const float*)d_in[16];
  const float* bp2  = (const float*)d_in[17];
  const float* gp2  = (const float*)d_in[18];
  const float* bbp2 = (const float*)d_in[19];
  const float* Wu1  = (const float*)d_in[20];
  const float* bu1  = (const float*)d_in[21];
  const float* gu1  = (const float*)d_in[22];
  const float* bbu1 = (const float*)d_in[23];
  const int* src = ei;            // edge_index[0]
  const int* dst = ei + NEDGES;   // edge_index[1]

  char* ws = (char*)d_ws;
  size_t off = 0;
  auto take = [&](size_t bytes) -> char* {
    char* p = ws + off;
    off += (bytes + 255) & ~(size_t)255;
    return p;
  };

  bf16_t* hbf  = (bf16_t*)take((size_t)NNODES * 128 * 2);
  bf16_t* WT1  = (bf16_t*)take(128 * 256 * 2);
  bf16_t* WT2  = (bf16_t*)take(128 * 128 * 2);
  bf16_t* WTp1 = (bf16_t*)take(128 * 128 * 2);
  bf16_t* WTu1 = (bf16_t*)take(128 * 256 * 2);
  // --- zero-initialized region start ---
  float* stats1 = (float*)take(256 * 4);
  float* stats2 = (float*)take(256 * 4);
  float* stats3 = (float*)take(256 * 4);
  float* stats5 = (float*)take(256 * 4);
  float* stats4 = (float*)take(2 * 4);
  float* haggr  = (float*)take((size_t)NNODES * 128 * 4);
  float* xsum   = (float*)take((size_t)NNODES * 2 * 4);
  float* cnt    = (float*)take((size_t)NNODES * 4);
  char* zend = ws + off;
  // --- zero-initialized region end ---
  float* ab1 = (float*)take(256 * 4);  // alpha[0..128) beta[128..256)
  float* ab2 = (float*)take(256 * 4);
  float* ab3 = (float*)take(256 * 4);
  float* ab5 = (float*)take(256 * 4);
  float* ab4 = (float*)take(2 * 4);
  float* tpre = (float*)take((size_t)NEDGES * 4);
  bf16_t* A1  = (bf16_t*)take((size_t)NEDGES * 128 * 2);  // A1, then reused as S1
  bf16_t* Msg = (bf16_t*)take((size_t)NEDGES * 128 * 2);  // msg_pre, then reused as Upre
  bf16_t* S1 = A1;
  bf16_t* Upre = Msg;

  long nz = (long)((zend - (char*)stats1) / 4);
  zero_kernel<<<2048, 256, 0, stream>>>(stats1, nz);

  cvt_bf16_kernel<<<(NNODES * 128 + 255) / 256, 256, 0, stream>>>(h, hbf, (long)NNODES * 128);
  transpose_cvt_kernel<<<(256 * 128 + 255) / 256, 256, 0, stream>>>(Wm1, WT1, 256, 128);
  transpose_cvt_kernel<<<(128 * 128 + 255) / 256, 256, 0, stream>>>(Wm2, WT2, 128, 128);
  transpose_cvt_kernel<<<(128 * 128 + 255) / 256, 256, 0, stream>>>(Wp1, WTp1, 128, 128);
  transpose_cvt_kernel<<<(256 * 128 + 255) / 256, 256, 0, stream>>>(Wu1, WTu1, 256, 128);

  int egrid = (ETILES + WPB - 1) / WPB;   // 3907
  int ngrid = (NODETILES + WPB - 1) / WPB; // 391

  edge_gemm1_kernel<<<egrid, 256, 0, stream>>>(hbf, src, dst, pos, WT1, Wm1 + 256 * 128, bm1,
                                               A1, stats1, ETILES);
  bn_finalize_kernel<<<1, 128, 0, stream>>>(stats1, gm1, bbm1, (float)NEDGES, 128, ab1, ab1 + 128);

  edge_gemm2_kernel<<<egrid, 256, 0, stream>>>(A1, ab1, ab1 + 128, WT2, bm2, Msg, stats2, ETILES);
  bn_finalize_kernel<<<1, 128, 0, stream>>>(stats2, gm2, bbm2, (float)NEDGES, 128, ab2, ab2 + 128);

  edge_gemm3_kernel<<<egrid, 256, 0, stream>>>(Msg, ab2, ab2 + 128, WTp1, bp1, dst, haggr, cnt,
                                               S1, stats3, ETILES);
  bn_finalize_kernel<<<1, 128, 0, stream>>>(stats3, gp1, bbp1, (float)NEDGES, 128, ab3, ab3 + 128);

  matvec_p2_kernel<<<(NEDGES + 255) / 256, 256, 0, stream>>>(S1, ab3, ab3 + 128, Wp2, bp2,
                                                             tpre, stats4, NEDGES);
  bn_finalize_kernel<<<1, 128, 0, stream>>>(stats4, gp2, bbp2, (float)NEDGES, 1, ab4, ab4 + 1);

  pos_scatter_kernel<<<(NEDGES + 255) / 256, 256, 0, stream>>>(tpre, ab4, src, dst, pos, xsum, NEDGES);

  node_gemm_kernel<<<ngrid, 256, 0, stream>>>(hbf, haggr, WTu1, bu1, Upre, stats5, NODETILES);
  bn_finalize_kernel<<<1, 128, 0, stream>>>(stats5, gu1, bbu1, (float)NNODES, 128, ab5, ab5 + 128);

  final_h_kernel<<<(NNODES * 128 + 255) / 256, 256, 0, stream>>>(h, Upre, ab5, ab5 + 128,
                                                                 (float*)d_out, (long)NNODES * 128);
  final_pos_kernel<<<(NNODES * 2 + 255) / 256, 256, 0, stream>>>(pos, xsum, cnt,
                                                                 (float*)d_out + (size_t)NNODES * 128,
                                                                 NNODES * 2);
}
